// ArcFaceInnerProduct_77369540870225
// MI455X (gfx1250) — matrix operand order repaired and checked by reading, not compiled, not run on hardware
//
#include <hip/hip_runtime.h>
#include <hip/hip_bf16.h>
#include <math.h>

// ---------------------------------------------------------------------------
// ArcFace inner product for MI455X (gfx1250, wave32, WMMA).
// feat [256,512] f32, weights [100000,512] f32, label [256] i32.
// Outputs (concat): marginal_logits [256,100000], 30*cos [256,100000],
//                   weights copy [100000,512].
// Strategy: split-f16 GEMM (hi+lo) with v_wmma_f32_16x16x32_f16, f32 accum.
// ---------------------------------------------------------------------------

typedef __attribute__((ext_vector_type(16))) _Float16 v16h;
typedef __attribute__((ext_vector_type(8)))  _Float16 v8h;
typedef __attribute__((ext_vector_type(8)))  float    v8f;

union HFrag { v16h v; v8h h[2]; };
union FAcc  { v8f  v; float f[8]; };

#define ARC_B 256
#define ARC_D 512
#define ARC_C 100000
#define ARC_CP 100032   // C padded to multiple of 64

__device__ __forceinline__ float blockSumSq256(float ss) {
  // 256 threads = 8 waves of 32
  #pragma unroll
  for (int off = 16; off > 0; off >>= 1) ss += __shfl_xor(ss, off, 32);
  __shared__ float part[8];
  const int t = threadIdx.x;
  if ((t & 31) == 0) part[t >> 5] = ss;
  __syncthreads();
  float tot = 0.f;
  #pragma unroll
  for (int i = 0; i < 8; ++i) tot += part[i];
  return tot;
}

// --- Kernel 1: normalize feat rows -> hi/lo f16 ----------------------------
__global__ __launch_bounds__(256) void arcface_norm_feat(
    const float* __restrict__ feat, _Float16* __restrict__ hi,
    _Float16* __restrict__ lo) {
  const int r = blockIdx.x;
  const int t = threadIdx.x;
  const float* row = feat + (size_t)r * ARC_D;
  const float x0 = row[t];
  const float x1 = row[t + 256];
  const float tot = blockSumSq256(x0 * x0 + x1 * x1);
  const float rn = 1.0f / sqrtf(tot);
  const float v0 = x0 * rn, v1 = x1 * rn;
  const _Float16 h0 = (_Float16)v0, h1 = (_Float16)v1;
  const _Float16 l0 = (_Float16)(v0 - (float)h0);
  const _Float16 l1 = (_Float16)(v1 - (float)h1);
  const size_t base = (size_t)r * ARC_D;
  hi[base + t] = h0;       hi[base + t + 256] = h1;
  lo[base + t] = l0;       lo[base + t + 256] = l1;
}

// --- Kernel 2: normalize weight rows -> hi/lo f16, copy raw to out ---------
__global__ __launch_bounds__(256) void arcface_norm_w(
    const float* __restrict__ w, _Float16* __restrict__ hi,
    _Float16* __restrict__ lo, float* __restrict__ wcopy) {
  const int r = blockIdx.x;
  const int t = threadIdx.x;
  const size_t base = (size_t)r * ARC_D;
  if (r < ARC_C) {
    const float x0 = w[base + t];
    const float x1 = w[base + t + 256];
    const float tot = blockSumSq256(x0 * x0 + x1 * x1);
    const float rn = 1.0f / sqrtf(tot);
    const float v0 = x0 * rn, v1 = x1 * rn;
    const _Float16 h0 = (_Float16)v0, h1 = (_Float16)v1;
    const _Float16 l0 = (_Float16)(v0 - (float)h0);
    const _Float16 l1 = (_Float16)(v1 - (float)h1);
    hi[base + t] = h0;       hi[base + t + 256] = h1;
    lo[base + t] = l0;       lo[base + t + 256] = l1;
    __builtin_nontemporal_store(x0, wcopy + base + t);
    __builtin_nontemporal_store(x1, wcopy + base + t + 256);
  } else {
    // zero padding rows so GEMM loads are clean
    hi[base + t] = (_Float16)0.f; hi[base + t + 256] = (_Float16)0.f;
    lo[base + t] = (_Float16)0.f; lo[base + t + 256] = (_Float16)0.f;
  }
}

// --- Kernel 3: split-f16 WMMA GEMM + ArcFace margin epilogue ---------------
// Block: 256 threads = 8 waves. Block tile: M=256 (all rows) x N=64.
// Wave w: rows [32w, 32w+32) as two 16-tiles; 4 N-tiles of 16 (shared N range).
__global__ __launch_bounds__(256, 1) void arcface_wmma_gemm(
    const _Float16* __restrict__ nf_hi, const _Float16* __restrict__ nf_lo,
    const _Float16* __restrict__ nw_hi, const _Float16* __restrict__ nw_lo,
    const int* __restrict__ label, float* __restrict__ out_marg,
    float* __restrict__ out_cos) {
  const int lane = threadIdx.x & 31;
  const int wv   = threadIdx.x >> 5;
  const int g    = lane >> 4;   // half-wave group
  const int ln   = lane & 15;
  const int  mbase = wv << 5;                   // 2 M-tiles: mbase, mbase+16
  const long nbase = (long)blockIdx.x << 6;     // 4 N-tiles

  // A fragment base pointers (16-bit A 16x32 layout: lane holds row M=ln,
  // halfs at K = kb+8g+[0..7] and kb+8g+16+[0..7])
  const _Float16* pah[2];
  const _Float16* pal[2];
  pah[0] = nf_hi + (size_t)(mbase + ln) * ARC_D + 8 * g;
  pah[1] = nf_hi + (size_t)(mbase + 16 + ln) * ARC_D + 8 * g;
  pal[0] = nf_lo + (size_t)(mbase + ln) * ARC_D + 8 * g;
  pal[1] = nf_lo + (size_t)(mbase + 16 + ln) * ARC_D + 8 * g;

  // B fragment base pointers (16-bit B 32x16 layout: lane holds col N=ln,
  // halfs at K = kb+8g+[0..7] and kb+8g+16+[0..7])
  const _Float16* pbh[4];
  const _Float16* pbl[4];
  #pragma unroll
  for (int nt = 0; nt < 4; ++nt) {
    const size_t cls = (size_t)(nbase + nt * 16 + ln);
    pbh[nt] = nw_hi + cls * ARC_D + 8 * g;
    pbl[nt] = nw_lo + cls * ARC_D + 8 * g;
  }

  const v8f zacc = {};
  v8f acc[2][4];
  #pragma unroll
  for (int mt = 0; mt < 2; ++mt)
    #pragma unroll
    for (int nt = 0; nt < 4; ++nt) acc[mt][nt] = zacc;

  #pragma unroll 1
  for (int kb = 0; kb < ARC_D; kb += 32) {
    HFrag a_hi[2], a_lo[2], b_hi[4], b_lo[4];
    #pragma unroll
    for (int mt = 0; mt < 2; ++mt) {
      a_hi[mt].h[0] = *(const v8h*)(pah[mt] + kb);
      a_hi[mt].h[1] = *(const v8h*)(pah[mt] + kb + 16);
      a_lo[mt].h[0] = *(const v8h*)(pal[mt] + kb);
      a_lo[mt].h[1] = *(const v8h*)(pal[mt] + kb + 16);
    }
    #pragma unroll
    for (int nt = 0; nt < 4; ++nt) {
      b_hi[nt].h[0] = __builtin_nontemporal_load((const v8h*)(pbh[nt] + kb));
      b_hi[nt].h[1] = __builtin_nontemporal_load((const v8h*)(pbh[nt] + kb + 16));
      b_lo[nt].h[0] = __builtin_nontemporal_load((const v8h*)(pbl[nt] + kb));
      b_lo[nt].h[1] = __builtin_nontemporal_load((const v8h*)(pbl[nt] + kb + 16));
      if (kb + 32 < ARC_D) {
        __builtin_prefetch(pbh[nt] + kb + 32, 0, 0);
        __builtin_prefetch(pbl[nt] + kb + 32, 0, 0);
      }
    }
    #pragma unroll
    for (int mt = 0; mt < 2; ++mt) {
      #pragma unroll
      for (int nt = 0; nt < 4; ++nt) {
        v8f c = acc[mt][nt];
        c = __builtin_amdgcn_wmma_f32_16x16x32_f16(
            false, a_hi[mt].v, false, b_hi[nt].v, (short)0, c, false, false);
        c = __builtin_amdgcn_wmma_f32_16x16x32_f16(
            false, a_hi[mt].v, false, b_lo[nt].v, (short)0, c, false, false);
        c = __builtin_amdgcn_wmma_f32_16x16x32_f16(
            false, a_lo[mt].v, false, b_hi[nt].v, (short)0, c, false, false);
        acc[mt][nt] = c;
      }
    }
  }

  // Epilogue: D layout = VGPR i, lanes 0-15 -> M=i, lanes 16-31 -> M=i+8.
  const float SCALE = 30.0f;
  const float COS_M = 0.87758256189037271612f;   // cos(0.5)
  const float SIN_M = 0.47942553860420300027f;   // sin(0.5)
  const float THRESH = -COS_M;
  const float MMSIN = 0.5f * SIN_M;              // margin * sin(margin)

  #pragma unroll
  for (int mt = 0; mt < 2; ++mt) {
    float cb[4][8];
    #pragma unroll
    for (int nt = 0; nt < 4; ++nt) {
      FAcc fa; fa.v = acc[mt][nt];
      #pragma unroll
      for (int i = 0; i < 8; ++i) cb[nt][i] = fa.f[i];
    }
    #pragma unroll
    for (int i = 0; i < 8; ++i) {
      const int r = mbase + mt * 16 + i + 8 * g;
      const int lab = label[r];
      const size_t rowoff = (size_t)r * ARC_C;
      #pragma unroll
      for (int nt = 0; nt < 4; ++nt) {
        const long c = nbase + nt * 16 + ln;
        if (c < ARC_C) {
          const float cosv = cb[nt][i];
          const float sc = SCALE * cosv;
          __builtin_nontemporal_store(sc, out_cos + rowoff + (size_t)c);
          float marg = sc;
          if ((int)c == lab) {
            if (cosv > THRESH) {
              const float s = sqrtf(fmaxf(1.0f - cosv * cosv, 0.0f));
              marg = SCALE * (cosv * COS_M - s * SIN_M);
            } else {
              marg = SCALE * (cosv - MMSIN);
            }
          }
          __builtin_nontemporal_store(marg, out_marg + rowoff + (size_t)c);
        }
      }
    }
  }
}

extern "C" void kernel_launch(void* const* d_in, const int* in_sizes, int n_in,
                              void* d_out, int out_size, void* d_ws, size_t ws_size,
                              hipStream_t stream) {
  (void)in_sizes; (void)n_in; (void)out_size; (void)ws_size;
  const float* feat    = (const float*)d_in[0];
  const float* weights = (const float*)d_in[1];
  const int*   label   = (const int*)d_in[2];

  float* out      = (float*)d_out;
  float* out_marg = out;                                      // [B,C]
  float* out_cos  = out + (size_t)ARC_B * ARC_C;              // [B,C]
  float* out_w    = out + (size_t)2 * ARC_B * ARC_C;          // [C,D]

  // Workspace layout (all 256B-aligned offsets)
  char* ws = (char*)d_ws;
  _Float16* nf_hi = (_Float16*)(ws);
  _Float16* nf_lo = (_Float16*)(ws + (size_t)ARC_B * ARC_D * 2);
  _Float16* nw_hi = (_Float16*)(ws + (size_t)2 * ARC_B * ARC_D * 2);
  _Float16* nw_lo = (_Float16*)(ws + (size_t)2 * ARC_B * ARC_D * 2 +
                                (size_t)ARC_CP * ARC_D * 2);

  arcface_norm_feat<<<ARC_B, 256, 0, stream>>>(feat, nf_hi, nf_lo);
  arcface_norm_w<<<ARC_CP, 256, 0, stream>>>(weights, nw_hi, nw_lo, out_w);
  arcface_wmma_gemm<<<ARC_CP / 64, 256, 0, stream>>>(
      nf_hi, nf_lo, nw_hi, nw_lo, label, out_marg, out_cos);
}
